// RSCNNBackbone_39779987096281
// MI455X (gfx1250) — matrix operand order repaired and checked by reading, not compiled, not run on hardware
//
#include <hip/hip_runtime.h>
#include <hip/hip_bf16.h>

// ---------------------------------------------------------------------------
// CDNA5 (gfx1250) RSCNN backbone. wave32. All GEMMs via v_wmma_f32_16x16x32_bf16.
// Activations live in bf16 in A-operand order; weights pre-transposed bf16
// [cout][Kp] with channel permutation [feats, xyz, pad] folded in.
// ---------------------------------------------------------------------------

typedef __attribute__((ext_vector_type(16))) __bf16 v16bf;
typedef __attribute__((ext_vector_type(8)))  float  v8f;

#define DEV static __device__ __forceinline__

DEV unsigned short f2bf(float f) {
  union { float f; unsigned u; } x; x.f = f;
  unsigned u = x.u;
  u += 0x7fffu + ((u >> 16) & 1u);   // round-to-nearest-even
  return (unsigned short)(u >> 16);
}
DEV float bf2f(unsigned short s) {
  union { unsigned u; float f; } x;
  x.u = ((unsigned)s) << 16;
  return x.f;
}

union AFrag { uint4 q[2]; unsigned short s[16]; v16bf v; };
union BFrag { uint4 q[2]; v16bf v; };

// B operand (K=32 step): lane L -> column n = L&15; lo chunk k=[kk+half*8,+8),
// hi chunk k=[kk+16+half*8,+8). Transposed bf16 weights -> two 16B loads.
DEV v16bf load_b_frag(const unsigned short* __restrict__ Wt, int n, int Kp,
                      int kk, int half) {
  BFrag b;
  const unsigned short* p = Wt + (size_t)n * Kp + kk + half * 8;
  b.q[0] = *reinterpret_cast<const uint4*>(p);
  b.q[1] = *reinterpret_cast<const uint4*>(p + 16);
  return b.v;
}

// A operand (K=32 step): lane L -> row m = L&15; lo chunk k=[kk+half*8,+8),
// hi chunk k=[kk+16+half*8,+8). bf16 source -> two 16B loads.
DEV v16bf load_a_frag(const unsigned short* __restrict__ row, int kk,
                      int half) {
  AFrag a;
  a.q[0] = *reinterpret_cast<const uint4*>(row + kk + half * 8);
  a.q[1] = *reinterpret_cast<const uint4*>(row + kk + 16 + half * 8);
  return a.v;
}

DEV v8f wmma_bf16(v16bf a, v16bf b, v8f c) {
  return __builtin_amdgcn_wmma_f32_16x16x32_bf16(false, a, false, b, (short)0,
                                                 c, false, false);
}

// Build the xyz+pad tail fragment (channels [Cf, Cf+32) hold x,y,z,0,...).
DEV v16bf make_tail_frag(float rx, float ry, float rz, int half) {
  AFrag a;
#pragma unroll
  for (int e = 0; e < 16; ++e) {
    int kl = (e < 8) ? half * 8 + e : 16 + half * 8 + (e - 8);
    float v = (kl == 0) ? rx : (kl == 1) ? ry : (kl == 2) ? rz : 0.0f;
    a.s[e] = f2bf(v);
  }
  return a.v;
}

// ---------------------------------------------------------------------------
// Weight conversion: f32 [cin][cout] -> bf16 [cout][Kp], zero-padded.
// cxyz >= 0: permute channels to [feats(Cf=cxyz), xyz(3), pad].
// ---------------------------------------------------------------------------
__global__ void rscnn_convert_weight(const float* __restrict__ W,
                                     unsigned short* __restrict__ Wt, int cin,
                                     int cout, int Kp, int cxyz) {
  int t = blockIdx.x * blockDim.x + threadIdx.x;
  if (t >= cout * Kp) return;
  int n = t / Kp, k = t % Kp;
  int orig;
  if (cxyz < 0) orig = (k < cin) ? k : -1;
  else if (k < cxyz) orig = k + 3;
  else if (k - cxyz < 3) orig = k - cxyz;
  else orig = -1;
  float v = (orig >= 0 && orig < cin) ? W[(size_t)orig * cout + n] : 0.0f;
  Wt[(size_t)n * Kp + k] = f2bf(v);
}

// ---------------------------------------------------------------------------
// Farthest point sampling: one 1024-thread block per batch, points + dmin in
// registers, two-stage wave32 shuffle argmax, winner coords through LDS.
// ---------------------------------------------------------------------------
__global__ __launch_bounds__(1024) void rscnn_fps(
    const float* __restrict__ xyz, long xyz_bs, int N, int np,
    float* __restrict__ oxyz, long o_bs) {
  __shared__ float swv[32];
  __shared__ int swi[32];
  __shared__ float lx, ly, lz;
  __shared__ int lwin;
  int b = blockIdx.x;
  xyz += (size_t)b * xyz_bs;
  oxyz += (size_t)b * o_bs;
  int tid = threadIdx.x;
  const int P = 8;
  float px[P], py[P], pz[P], dm[P];
#pragma unroll
  for (int i = 0; i < P; ++i) {
    int p = tid + i * 1024;
    if (p < N) {
      px[i] = xyz[p * 3 + 0];
      py[i] = xyz[p * 3 + 1];
      pz[i] = xyz[p * 3 + 2];
      dm[i] = 1e10f;
    }
  }
  if (tid == 0) {
    lx = xyz[0]; ly = xyz[1]; lz = xyz[2];
    oxyz[0] = lx; oxyz[1] = ly; oxyz[2] = lz;
  }
  __syncthreads();
  for (int t = 1; t < np; ++t) {
    float bx = lx, by = ly, bz = lz;
    float bestv = -1.0f;
    int besti = 0x7fffffff;
#pragma unroll
    for (int i = 0; i < P; ++i) {
      int p = tid + i * 1024;
      if (p < N) {
        float dx = px[i] - bx, dy = py[i] - by, dz = pz[i] - bz;
        float d = dx * dx + dy * dy + dz * dz;
        if (d < dm[i]) dm[i] = d;
        float v = dm[i];
        if (v > bestv || (v == bestv && p < besti)) { bestv = v; besti = p; }
      }
    }
#pragma unroll
    for (int off = 16; off; off >>= 1) {
      float ov = __shfl_down(bestv, off);
      int oi = __shfl_down(besti, off);
      if (ov > bestv || (ov == bestv && oi < besti)) { bestv = ov; besti = oi; }
    }
    if ((tid & 31) == 0) { swv[tid >> 5] = bestv; swi[tid >> 5] = besti; }
    __syncthreads();
    if (tid < 32) {
      bestv = swv[tid];
      besti = swi[tid];
#pragma unroll
      for (int off = 16; off; off >>= 1) {
        float ov = __shfl_down(bestv, off);
        int oi = __shfl_down(besti, off);
        if (ov > bestv || (ov == bestv && oi < besti)) { bestv = ov; besti = oi; }
      }
      if (tid == 0) lwin = besti;
    }
    __syncthreads();
    int win = lwin;
    if ((win & 1023) == tid) {
      int i = win >> 10;
      lx = px[i]; ly = py[i]; lz = pz[i];
      oxyz[t * 3 + 0] = px[i];
      oxyz[t * 3 + 1] = py[i];
      oxyz[t * 3 + 2] = pz[i];
    }
    __syncthreads();
  }
}

// ---------------------------------------------------------------------------
// Ball query: one wave per query point; first ns in-radius indices in
// ascending order (wave ballot prefix scan), pad with first hit (or 0).
// ---------------------------------------------------------------------------
__global__ __launch_bounds__(32) void rscnn_ball_query(
    const float* __restrict__ xyz, long xyz_bs, int N,
    const float* __restrict__ ctr, long ctr_bs, float r2, int ns,
    int* __restrict__ idx, long idx_bs) {
  int s = blockIdx.x, b = blockIdx.z, lane = threadIdx.x;
  xyz += (size_t)b * xyz_bs;
  ctr += (size_t)b * ctr_bs;
  idx += (size_t)b * idx_bs + (size_t)s * ns;
  float cx = ctr[s * 3 + 0], cy = ctr[s * 3 + 1], cz = ctr[s * 3 + 2];
  int cnt = 0, first = -1;
  for (int base = 0; base < N && cnt < ns; base += 32) {
    int i = base + lane;
    bool hit = false;
    if (i < N) {
      float dx = xyz[i * 3 + 0] - cx, dy = xyz[i * 3 + 1] - cy,
            dz = xyz[i * 3 + 2] - cz;
      hit = (dx * dx + dy * dy + dz * dz) <= r2;
    }
    unsigned mm = (unsigned)__ballot(hit);
    int pos = __popc(mm & ((1u << lane) - 1u));
    if (hit && cnt + pos < ns) idx[cnt + pos] = i;
    int c = __popc(mm);
    if (first < 0 && c > 0) first = base + __ffs(mm) - 1;
    cnt += c;
    if (cnt > ns) cnt = ns;
  }
  int fill = (first < 0) ? 0 : first;
  for (int t = cnt + lane; t < ns; t += 32) idx[t] = fill;
}

// ---------------------------------------------------------------------------
// Grouped MLP + max over samples. One wave per (point s, 16-col tile).
// A rows gathered straight from bf16 feature rows (2x16B loads per K-step);
// xyz-relative channels live in the single tail block. Invalid rows are
// clamped (their garbage output is excluded by the max guard).
// ---------------------------------------------------------------------------
__global__ __launch_bounds__(32) void rscnn_sa_grouped_gemm(
    const float* __restrict__ xyz, long xyz_bs, const float* __restrict__ ctr,
    long ctr_bs, const int* __restrict__ idx, long idx_bs,
    const unsigned short* __restrict__ feats, long feats_bs, int Cf,
    const unsigned short* __restrict__ Wt, int Kp,
    const float* __restrict__ bias, unsigned short* __restrict__ fout,
    long fout_bs, int coutOff, int coutTotal, int ns) {
  int s = blockIdx.x, nT = blockIdx.y, b = blockIdx.z;
  int lane = threadIdx.x, half = lane >> 4, l15 = lane & 15;
  xyz += (size_t)b * xyz_bs;
  ctr += (size_t)b * ctr_bs;
  idx += (size_t)b * idx_bs;
  if (feats) feats += (size_t)b * feats_bs;
  fout += (size_t)b * fout_bs;
  float cx = ctr[s * 3 + 0], cy = ctr[s * 3 + 1], cz = ctr[s * 3 + 2];
  int ncol = nT * 16 + l15;
  float bval = bias[ncol];
  float gmax = 0.0f;
  int nMT = (ns + 15) >> 4;
  for (int mt = 0; mt < nMT; ++mt) {
    int j = mt * 16 + l15;
    int jc = j < ns ? j : ns - 1;       // clamp: garbage rows filtered by max
    int p = idx[(size_t)s * ns + jc];
    float rx = xyz[p * 3 + 0] - cx;
    float ry = xyz[p * 3 + 1] - cy;
    float rz = xyz[p * 3 + 2] - cz;
    v16bf atail = make_tail_frag(rx, ry, rz, half);
    const unsigned short* frow = feats ? feats + (size_t)p * Cf : nullptr;
    v8f acc = {};
    for (int kk = 0; kk < Cf; kk += 32)
      acc = wmma_bf16(load_a_frag(frow, kk, half),
                      load_b_frag(Wt, ncol, Kp, kk, half), acc);
    acc = wmma_bf16(atail, load_b_frag(Wt, ncol, Kp, Cf, half), acc);
    float cm = 0.0f;
#pragma unroll
    for (int r = 0; r < 8; ++r) {
      int row = mt * 16 + half * 8 + r;
      float h = acc[r] + bval;
      h = h > 0.f ? h : 0.f;
      if (row < ns && h > cm) cm = h;
    }
    if (cm > gmax) gmax = cm;
  }
  float other = __shfl_xor(gmax, 16);
  if (other > gmax) gmax = other;
  if (lane < 16)
    fout[(size_t)s * coutTotal + coutOff + nT * 16 + lane] = f2bf(gmax);
}

// ---------------------------------------------------------------------------
// Global pool MLP: relu(concat(f,xyz) @ W + b).max(points) -> f32 128-vector
// ---------------------------------------------------------------------------
__global__ __launch_bounds__(32) void rscnn_pool_gemm(
    const float* __restrict__ xyz, long xyz_bs,
    const unsigned short* __restrict__ feats, long f_bs, int Cf,
    const unsigned short* __restrict__ Wt, int Kp,
    const float* __restrict__ bias, float* __restrict__ gout, long g_bs,
    int Npts) {
  int nT = blockIdx.y, b = blockIdx.z;
  int lane = threadIdx.x, half = lane >> 4, l15 = lane & 15;
  xyz += (size_t)b * xyz_bs;
  feats += (size_t)b * f_bs;
  gout += (size_t)b * g_bs;
  int ncol = nT * 16 + l15;
  float bval = bias[ncol];
  float gmax = 0.0f;
  int nMT = (Npts + 15) >> 4;
  for (int mt = 0; mt < nMT; ++mt) {
    int j = mt * 16 + l15;
    int jc = j < Npts ? j : Npts - 1;
    v16bf atail =
        make_tail_frag(xyz[jc * 3 + 0], xyz[jc * 3 + 1], xyz[jc * 3 + 2], half);
    const unsigned short* frow = feats + (size_t)jc * Cf;
    v8f acc = {};
    for (int kk = 0; kk < Cf; kk += 32)
      acc = wmma_bf16(load_a_frag(frow, kk, half),
                      load_b_frag(Wt, ncol, Kp, kk, half), acc);
    acc = wmma_bf16(atail, load_b_frag(Wt, ncol, Kp, Cf, half), acc);
    float cm = 0.0f;
#pragma unroll
    for (int r = 0; r < 8; ++r) {
      int row = mt * 16 + half * 8 + r;
      float h = acc[r] + bval;
      h = h > 0.f ? h : 0.f;
      if (row < Npts && h > cm) cm = h;
    }
    if (cm > gmax) gmax = cm;
  }
  float other = __shfl_xor(gmax, 16);
  if (other > gmax) gmax = other;
  if (lane < 16) gout[nT * 16 + lane] = gmax;
}

// ---------------------------------------------------------------------------
// Dense GEMM: out = act(concat(src0,src1) @ Wt^T + bias).
// C0 is a multiple of 32, so each 32-K block reads from exactly one source:
// one branchless pointer select per WMMA. outKind: 0 = bf16 row-major,
// 1 = f32 transposed (final layer -> d_out[c][n]).
// ---------------------------------------------------------------------------
__global__ __launch_bounds__(32) void rscnn_dense_gemm(
    const unsigned short* __restrict__ src0, long s0_bs, int C0,
    const unsigned short* __restrict__ src1, long s1_bs, int C1,
    const unsigned short* __restrict__ Wt, int Kp,
    const float* __restrict__ bias, long bias_bs, void* __restrict__ out,
    long out_bs, int N, int cout, int relu, int outKind) {
  int mT = blockIdx.x, nT = blockIdx.y, b = blockIdx.z;
  int lane = threadIdx.x, half = lane >> 4, l15 = lane & 15;
  if (src0) src0 += (size_t)b * s0_bs;
  if (src1) src1 += (size_t)b * s1_bs;
  bias += (size_t)b * bias_bs;
  int m = mT * 16 + l15;
  int mc = m < N ? m : N - 1;           // clamp: rows >= N never stored
  const unsigned short* p0 = src0 + (size_t)mc * C0;
  const unsigned short* p1 = src1 + (size_t)mc * C1;
  int ncol = nT * 16 + l15;
  v8f acc = {};
  for (int kk = 0; kk < Kp; kk += 32) {
    const unsigned short* base = (kk < C0) ? (p0 + kk) : (p1 + (kk - C0));
    acc = wmma_bf16(load_a_frag(base, 0, half),
                    load_b_frag(Wt, ncol, Kp, kk, half), acc);
  }
  float bv = bias[ncol];
  if (outKind == 0) {
    unsigned short* ob = (unsigned short*)out + (size_t)b * out_bs;
#pragma unroll
    for (int r = 0; r < 8; ++r) {
      int row = mT * 16 + half * 8 + r;
      if (row < N) {
        float h = acc[r] + bv;
        if (relu) h = h > 0.f ? h : 0.f;
        ob[(size_t)row * cout + ncol] = f2bf(h);
      }
    }
  } else {
    float* of = (float*)out + (size_t)b * out_bs;
#pragma unroll
    for (int r = 0; r < 8; ++r) {
      int row = mT * 16 + half * 8 + r;
      if (row < N) {
        float h = acc[r] + bv;
        if (relu) h = h > 0.f ? h : 0.f;
        of[(size_t)ncol * N + row] = h;
      }
    }
  }
}

// ---------------------------------------------------------------------------
// 3-NN inverse-distance interpolation: one wave per unknown point.
// ---------------------------------------------------------------------------
DEV void ins3(float d, int i, float bd[3], int bi[3]) {
  if (d < bd[2]) {
    bd[2] = d; bi[2] = i;
    if (bd[2] < bd[1]) {
      float td = bd[1]; bd[1] = bd[2]; bd[2] = td;
      int ti = bi[1]; bi[1] = bi[2]; bi[2] = ti;
    }
    if (bd[1] < bd[0]) {
      float td = bd[0]; bd[0] = bd[1]; bd[1] = td;
      int ti = bi[0]; bi[0] = bi[1]; bi[1] = ti;
    }
  }
}

__global__ __launch_bounds__(32) void rscnn_knn3_interp(
    const float* __restrict__ uxyz, long u_bs, int Nu,
    const float* __restrict__ kxyz, long k_bs, int Nk,
    const unsigned short* __restrict__ kf, long kf_bs, int C,
    unsigned short* __restrict__ out, long o_bs) {
  int u = blockIdx.x, b = blockIdx.z, lane = threadIdx.x;
  uxyz += (size_t)b * u_bs;
  kxyz += (size_t)b * k_bs;
  kf += (size_t)b * kf_bs;
  out += (size_t)b * o_bs;
  float ux = uxyz[u * 3 + 0], uy = uxyz[u * 3 + 1], uz = uxyz[u * 3 + 2];
  float bd[3] = {3.4e38f, 3.4e38f, 3.4e38f};
  int bi[3] = {0, 0, 0};
  for (int i = lane; i < Nk; i += 32) {
    float dx = kxyz[i * 3 + 0] - ux, dy = kxyz[i * 3 + 1] - uy,
          dz = kxyz[i * 3 + 2] - uz;
    ins3(dx * dx + dy * dy + dz * dz, i, bd, bi);
  }
#pragma unroll
  for (int off = 16; off; off >>= 1) {
    float od[3];
    int oi[3];
#pragma unroll
    for (int t = 0; t < 3; ++t) {
      od[t] = __shfl_xor(bd[t], off);
      oi[t] = __shfl_xor(bi[t], off);
    }
#pragma unroll
    for (int t = 0; t < 3; ++t) ins3(od[t], oi[t], bd, bi);
  }
  float w0 = 1.0f / (fmaxf(bd[0], 0.f) + 1e-8f);
  float w1 = 1.0f / (fmaxf(bd[1], 0.f) + 1e-8f);
  float w2 = 1.0f / (fmaxf(bd[2], 0.f) + 1e-8f);
  float inv = 1.0f / (w0 + w1 + w2);
  w0 *= inv; w1 *= inv; w2 *= inv;
  const unsigned short* r0 = kf + (size_t)bi[0] * C;
  const unsigned short* r1 = kf + (size_t)bi[1] * C;
  const unsigned short* r2 = kf + (size_t)bi[2] * C;
  for (int c = lane; c < C; c += 32)
    out[(size_t)u * C + c] =
        f2bf(w0 * bf2f(r0[c]) + w1 * bf2f(r1[c]) + w2 * bf2f(r2[c]));
}

// gbias[c] = fc_b[c] + g1 . Wfc[128:256,c] + g2 . Wfc[256:384,c]
__global__ void rscnn_gbias(const float* __restrict__ fcW,
                            const float* __restrict__ fcB,
                            const float* __restrict__ g1v,
                            const float* __restrict__ g2v,
                            float* __restrict__ gb, long v_bs) {
  int b = blockIdx.x, c = threadIdx.x;
  g1v += (size_t)b * v_bs;
  g2v += (size_t)b * v_bs;
  gb += (size_t)b * v_bs;
  float acc = fcB[c];
  for (int k = 0; k < 128; ++k) acc += g1v[k] * fcW[(size_t)(128 + k) * 128 + c];
  for (int k = 0; k < 128; ++k) acc += g2v[k] * fcW[(size_t)(256 + k) * 128 + c];
  gb[c] = acc;
}

// ---------------------------------------------------------------------------
// Host side
// ---------------------------------------------------------------------------
extern "C" void kernel_launch(void* const* d_in, const int* in_sizes, int n_in,
                              void* d_out, int out_size, void* d_ws,
                              size_t ws_size, hipStream_t stream) {
  (void)out_size; (void)ws_size;
  const int B = 4;

  // ---- identify inputs by group-leader element counts (order-robust) ----
  const float* pc = nullptr;
  const float *saW[4][3] = {}, *saB[4][3] = {};
  const float *gW[2] = {}, *gB[2] = {};
  const float *fpW[4][2] = {}, *fpB[4][2] = {};
  const float *fcW = nullptr, *fcB = nullptr;
  int i = 0;
  while (i < n_in) {
    int sz = in_sizes[i];
    if (sz == 98304 && !pc) { pc = (const float*)d_in[i]; i += 1; }
    else if (sz == 192) { for (int s = 0; s < 3; ++s) { saW[0][s] = (const float*)d_in[i + s]; saB[0][s] = (const float*)d_in[i + 3 + s]; } i += 6; }
    else if (sz == 24960) { for (int s = 0; s < 3; ++s) { saW[1][s] = (const float*)d_in[i + s]; saB[1][s] = (const float*)d_in[i + 3 + s]; } i += 6; }
    else if (sz == 99072) { for (int s = 0; s < 3; ++s) { saW[2][s] = (const float*)d_in[i + s]; saB[2][s] = (const float*)d_in[i + 3 + s]; } i += 6; }
    else if (sz == 394752) { for (int s = 0; s < 3; ++s) { saW[3][s] = (const float*)d_in[i + s]; saB[3][s] = (const float*)d_in[i + 3 + s]; } i += 6; }
    else if (sz == 196992) { gW[0] = (const float*)d_in[i]; gB[0] = (const float*)d_in[i + 1]; i += 2; }
    else if (sz == 98688) { gW[1] = (const float*)d_in[i]; gB[1] = (const float*)d_in[i + 1]; i += 2; }
    else if (sz == 32768) { fpW[0][0] = (const float*)d_in[i]; fpW[0][1] = (const float*)d_in[i + 1]; fpB[0][0] = (const float*)d_in[i + 2]; fpB[0][1] = (const float*)d_in[i + 3]; i += 4; }
    else if (sz == 180224) { fpW[1][0] = (const float*)d_in[i]; fpW[1][1] = (const float*)d_in[i + 1]; fpB[1][0] = (const float*)d_in[i + 2]; fpB[1][1] = (const float*)d_in[i + 3]; i += 4; }
    else if (sz == 458752) { fpW[2][0] = (const float*)d_in[i]; fpW[2][1] = (const float*)d_in[i + 1]; fpB[2][0] = (const float*)d_in[i + 2]; fpB[2][1] = (const float*)d_in[i + 3]; i += 4; }
    else if (sz == 1179648) { fpW[3][0] = (const float*)d_in[i]; fpW[3][1] = (const float*)d_in[i + 1]; fpB[3][0] = (const float*)d_in[i + 2]; fpB[3][1] = (const float*)d_in[i + 3]; i += 4; }
    else if (sz == 49152) { fcW = (const float*)d_in[i]; fcB = (const float*)d_in[i + 1]; i += 2; }
    else i += 1;
  }
  if (!pc || !fcW || !saW[0][0] || !fpW[3][0] || !gW[0]) return;

  // ---- carve workspace ----
  char* ws = (char*)d_ws;
  size_t off = 0;
  auto alloc = [&](size_t bytes) -> char* {
    off = (off + 255) & ~(size_t)255;
    char* p = ws + off;
    off += bytes;
    return p;
  };
  auto fbuf = [&](size_t elemsPerBatch) -> float* {
    return (float*)alloc((size_t)B * elemsPerBatch * 4);
  };
  auto ibuf = [&](size_t elemsPerBatch) -> int* {
    return (int*)alloc((size_t)B * elemsPerBatch * 4);
  };
  auto hbuf = [&](size_t elemsPerBatch) -> unsigned short* {
    return (unsigned short*)alloc((size_t)B * elemsPerBatch * 2);
  };
  auto wbuf = [&](size_t elems) -> unsigned short* {
    return (unsigned short*)alloc(elems * 2);
  };

  float* xyz1 = fbuf(2048 * 3);
  float* xyz2 = fbuf(1024 * 3);
  float* xyz3 = fbuf(256 * 3);
  float* xyz4 = fbuf(64 * 3);
  float* g1v = fbuf(128);
  float* g2v = fbuf(128);
  float* gb = fbuf(128);
  unsigned short* f1 = hbuf(2048 * 192);
  unsigned short* f2 = hbuf(1024 * 384);
  unsigned short* f3sa = hbuf(256 * 768);
  unsigned short* f4 = hbuf(64 * 1536);
  unsigned short* it3 = hbuf(256 * 1536);
  unsigned short* h3 = hbuf(256 * 512);
  unsigned short* f3n = hbuf(256 * 512);
  unsigned short* it2 = hbuf(1024 * 512);
  unsigned short* h2 = hbuf(1024 * 512);
  unsigned short* f2n = hbuf(1024 * 512);
  unsigned short* it1 = hbuf(2048 * 512);
  unsigned short* h1 = hbuf(2048 * 256);
  unsigned short* f1n = hbuf(2048 * 256);
  unsigned short* it0 = hbuf(8192 * 256);
  unsigned short* h0 = hbuf(8192 * 128);
  unsigned short* f0 = hbuf(8192 * 128);

  const int saKp[4] = {32, 224, 416, 800};
  const int saCin[4] = {3, 195, 387, 771};
  const int saCout[4] = {64, 128, 256, 512};
  int* lidx[4];
  const long lidx_bs[4] = {2048L * 96, 1024L * 128, 256L * 96, 64L * 72};
  for (int l = 0; l < 4; ++l) lidx[l] = ibuf((size_t)lidx_bs[l]);

  unsigned short* wt_sa[4][3];
  for (int l = 0; l < 4; ++l)
    for (int s = 0; s < 3; ++s)
      wt_sa[l][s] = wbuf((size_t)saCout[l] * saKp[l]);
  unsigned short* wt_g0 = wbuf(128 * 1568);
  unsigned short* wt_g1 = wbuf(128 * 800);
  unsigned short* wt_fp[4][2];
  const int fpKp[4][2] = {{256, 128}, {704, 256}, {896, 512}, {2304, 512}};
  const int fpCout[4][2] = {{128, 128}, {256, 256}, {512, 512}, {512, 512}};
  for (int l = 0; l < 4; ++l)
    for (int s = 0; s < 2; ++s)
      wt_fp[l][s] = wbuf((size_t)fpCout[l][s] * fpKp[l][s]);
  unsigned short* wt_fc = wbuf(128 * 128);

  // ---- weight conversion (f32 -> transposed bf16; SA/pool get the
  //      [feats, xyz, pad] channel permutation; dense layers none) ----
  struct Conv { const float* W; int cin, cout, Kp, cxyz; unsigned short* out; };
  Conv cds[23];
  int nc = 0;
  const int lCf[4] = {0, 192, 384, 768};
  for (int l = 0; l < 4; ++l)
    for (int s = 0; s < 3; ++s)
      cds[nc++] = {saW[l][s], saCin[l], saCout[l], saKp[l], lCf[l], wt_sa[l][s]};
  cds[nc++] = {gW[0], 1539, 128, 1568, 1536, wt_g0};
  cds[nc++] = {gW[1], 771, 128, 800, 768, wt_g1};
  for (int l = 0; l < 4; ++l)
    for (int s = 0; s < 2; ++s)
      cds[nc++] = {fpW[l][s], fpKp[l][s], fpCout[l][s], fpKp[l][s], -1,
                   wt_fp[l][s]};
  cds[nc++] = {fcW, 128, 128, 128, -1, wt_fc};
  for (int c = 0; c < nc; ++c) {
    int total = cds[c].cout * cds[c].Kp;
    rscnn_convert_weight<<<(total + 255) / 256, 256, 0, stream>>>(
        cds[c].W, cds[c].out, cds[c].cin, cds[c].cout, cds[c].Kp, cds[c].cxyz);
  }

  // ---- FPS chain ----
  rscnn_fps<<<B, 1024, 0, stream>>>(pc, 24576, 8192, 2048, xyz1, 6144);
  rscnn_fps<<<B, 1024, 0, stream>>>(xyz1, 6144, 2048, 1024, xyz2, 3072);
  rscnn_fps<<<B, 1024, 0, stream>>>(xyz2, 3072, 1024, 256, xyz3, 768);
  rscnn_fps<<<B, 1024, 0, stream>>>(xyz3, 768, 256, 64, xyz4, 192);

  // ---- SA levels ----
  const float rad[4][3] = {{20, 30, 40}, {40, 60, 80}, {80, 120, 160}, {200, 240, 280}};
  const int nsamp[4][3] = {{16, 32, 48}, {16, 48, 64}, {16, 32, 48}, {16, 24, 32}};
  const float* lx_xyz[4] = {pc, xyz1, xyz2, xyz3};
  const long lx_bs[4] = {24576, 6144, 3072, 768};
  const float* lc_xyz[4] = {xyz1, xyz2, xyz3, xyz4};
  const long lc_bs[4] = {6144, 3072, 768, 192};
  const unsigned short* lfeat[4] = {nullptr, f1, f2, f3sa};
  const long lfeat_bs[4] = {0, 2048L * 192, 1024L * 384, 256L * 768};
  unsigned short* lfout[4] = {f1, f2, f3sa, f4};
  const long lfout_bs[4] = {2048L * 192, 1024L * 384, 256L * 768, 64L * 1536};
  const int lS[4] = {2048, 1024, 256, 64};
  const int lN[4] = {8192, 2048, 1024, 256};
  const int lCoutTot[4] = {192, 384, 768, 1536};

  for (int l = 0; l < 4; ++l) {
    int soff = 0;
    for (int s = 0; s < 3; ++s) {
      int ns = nsamp[l][s];
      float r = rad[l][s];
      int* idxp = lidx[l] + soff;
      rscnn_ball_query<<<dim3(lS[l], 1, B), 32, 0, stream>>>(
          lx_xyz[l], lx_bs[l], lN[l], lc_xyz[l], lc_bs[l], r * r, ns, idxp,
          lidx_bs[l]);
      rscnn_sa_grouped_gemm<<<dim3(lS[l], saCout[l] / 16, B), 32, 0, stream>>>(
          lx_xyz[l], lx_bs[l], lc_xyz[l], lc_bs[l], idxp, lidx_bs[l], lfeat[l],
          lfeat_bs[l], lCf[l], wt_sa[l][s], saKp[l], saB[l][s], lfout[l],
          lfout_bs[l], s * saCout[l], lCoutTot[l], ns);
      soff += lS[l] * ns;
    }
  }

  // ---- global pools + fused FC bias ----
  rscnn_pool_gemm<<<dim3(1, 8, B), 32, 0, stream>>>(
      xyz4, 192, f4, 64L * 1536, 1536, wt_g0, 1568, gB[0], g1v, 128, 64);
  rscnn_pool_gemm<<<dim3(1, 8, B), 32, 0, stream>>>(
      xyz3, 768, f3sa, 256L * 768, 768, wt_g1, 800, gB[1], g2v, 128, 256);
  rscnn_gbias<<<B, 128, 0, stream>>>(fcW, fcB, g1v, g2v, gb, 128);

  // ---- feature propagation ----
  rscnn_knn3_interp<<<dim3(256, 1, B), 32, 0, stream>>>(
      xyz3, 768, 256, xyz4, 192, 64, f4, 64L * 1536, 1536, it3, 256L * 1536);
  rscnn_dense_gemm<<<dim3(16, 32, B), 32, 0, stream>>>(
      f3sa, 256L * 768, 768, it3, 256L * 1536, 1536, wt_fp[3][0], 2304,
      fpB[3][0], 0, h3, 256L * 512, 256, 512, 1, 0);
  rscnn_dense_gemm<<<dim3(16, 32, B), 32, 0, stream>>>(
      h3, 256L * 512, 512, h3, 256L * 512, 0, wt_fp[3][1], 512, fpB[3][1], 0,
      f3n, 256L * 512, 256, 512, 1, 0);

  rscnn_knn3_interp<<<dim3(1024, 1, B), 32, 0, stream>>>(
      xyz2, 3072, 1024, xyz3, 768, 256, f3n, 256L * 512, 512, it2, 1024L * 512);
  rscnn_dense_gemm<<<dim3(64, 32, B), 32, 0, stream>>>(
      f2, 1024L * 384, 384, it2, 1024L * 512, 512, wt_fp[2][0], 896, fpB[2][0],
      0, h2, 1024L * 512, 1024, 512, 1, 0);
  rscnn_dense_gemm<<<dim3(64, 32, B), 32, 0, stream>>>(
      h2, 1024L * 512, 512, h2, 1024L * 512, 0, wt_fp[2][1], 512, fpB[2][1], 0,
      f2n, 1024L * 512, 1024, 512, 1, 0);

  rscnn_knn3_interp<<<dim3(2048, 1, B), 32, 0, stream>>>(
      xyz1, 6144, 2048, xyz2, 3072, 1024, f2n, 1024L * 512, 512, it1,
      2048L * 512);
  rscnn_dense_gemm<<<dim3(128, 16, B), 32, 0, stream>>>(
      f1, 2048L * 192, 192, it1, 2048L * 512, 512, wt_fp[1][0], 704, fpB[1][0],
      0, h1, 2048L * 256, 2048, 256, 1, 0);
  rscnn_dense_gemm<<<dim3(128, 16, B), 32, 0, stream>>>(
      h1, 2048L * 256, 256, h1, 2048L * 256, 0, wt_fp[1][1], 256, fpB[1][1], 0,
      f1n, 2048L * 256, 2048, 256, 1, 0);

  rscnn_knn3_interp<<<dim3(8192, 1, B), 32, 0, stream>>>(
      pc, 24576, 8192, xyz1, 6144, 2048, f1n, 2048L * 256, 256, it0,
      8192L * 256);
  rscnn_dense_gemm<<<dim3(512, 8, B), 32, 0, stream>>>(
      it0, 8192L * 256, 0, it0, 8192L * 256, 256, wt_fp[0][0], 256, fpB[0][0],
      0, h0, 8192L * 128, 8192, 128, 1, 0);
  rscnn_dense_gemm<<<dim3(512, 8, B), 32, 0, stream>>>(
      h0, 8192L * 128, 128, h0, 8192L * 128, 0, wt_fp[0][1], 128, fpB[0][1], 0,
      f0, 8192L * 128, 8192, 128, 1, 0);

  // ---- final FC (globals folded into bias), transposed f32 store ----
  rscnn_dense_gemm<<<dim3(512, 8, B), 32, 0, stream>>>(
      f0, 8192L * 128, 128, f0, 8192L * 128, 0, wt_fc, 128, gb, 128, d_out,
      128L * 8192, 8192, 128, 0, 1);
}
